// ElectrodePositionAttention_14207751815441
// MI455X (gfx1250) — matrix operand order repaired and checked by reading, not compiled, not run on hardware
//
#include <hip/hip_runtime.h>

// ---------------------------------------------------------------------------
// CDNA5 (gfx1250) WMMA helpers, wave32.
// f16 16x16x32:  D(f32 16x16) = A(f16 16x32) * B(f16 32x16) + C
// A-frag layout (ISA 7.12.2): lane l: m = l&15, half = l>>4;
//   VGPR v holds k = ((v&4)?16:0) + half*8 + (v&3)*2  (pair k,k+1)
// B-frag layout (ISA 7.12.4/5 pattern): lane l: n = l&15, half = l>>4;
//   VGPR v holds k = ((v&4)?16:0) + half*8 + (v&3)*2 (pair k,k+1)   [requires n-major LDS tile]
// C/D f32: lane l: n = l&15; VGPR r: m = r + 8*(l>>4)
// ---------------------------------------------------------------------------
typedef __attribute__((ext_vector_type(16))) _Float16 v16h;
typedef __attribute__((ext_vector_type(8)))  float    v8f;

union FragU { v16h v; unsigned int u[8]; };

__device__ __forceinline__ v16h frag_a(const _Float16* smem, int stride) {
    const int lane = threadIdx.x & 31;
    const int m  = lane & 15;
    const int hf = lane >> 4;
    FragU f;
#pragma unroll
    for (int r = 0; r < 8; ++r) {
        const int kk = ((r & 4) ? 16 : 0) + hf * 8 + (r & 3) * 2;
        f.u[r] = *(const unsigned int*)(smem + m * stride + kk);
    }
    return f.v;
}

__device__ __forceinline__ v16h frag_b(const _Float16* smemT, int stride) {
    const int lane = threadIdx.x & 31;
    const int n  = lane & 15;
    const int hf = lane >> 4;
    FragU f;
#pragma unroll
    for (int r = 0; r < 8; ++r) {
        const int kk = ((r & 4) ? 16 : 0) + hf * 8 + (r & 3) * 2;
        f.u[r] = *(const unsigned int*)(smemT + n * stride + kk);
    }
    return f.v;
}

__device__ __forceinline__ v8f wmma32(v16h a, v16h b, v8f c) {
    return __builtin_amdgcn_wmma_f32_16x16x32_f16(false, a, false, b, (short)0, c,
                                                  false, false);
}

// ---------------------------------------------------------------------------
// CDNA5 async global->LDS copy (ASYNCcnt-tracked, ISA 10. / 15.18.3).
// LDS address comes from the VDST VGPR; for generic pointers to __shared__,
// LDS offset = addr[31:0] (flat aperture rule).
// ---------------------------------------------------------------------------
__device__ __forceinline__ void async_copy_b128(const _Float16* g, _Float16* l) {
    const unsigned lds = (unsigned)(size_t)l;
    asm volatile("global_load_async_to_lds_b128 %0, %1, off"
                 :: "v"(lds), "v"((unsigned long long)(size_t)g)
                 : "memory");
}
__device__ __forceinline__ void async_copy_row32(const _Float16* g, _Float16* l) {
#pragma unroll
    for (int j = 0; j < 4; ++j) async_copy_b128(g + j * 8, l + j * 8);
}
__device__ __forceinline__ void wait_async() {
    asm volatile("s_wait_asynccnt 0" ::: "memory");
}

// ---------------------------------------------------------------------------
// f32 -> f16 flat conversion (n multiple of 4)
// ---------------------------------------------------------------------------
__global__ __launch_bounds__(256) void cvt_f32_f16(const float* __restrict__ in,
                                                   _Float16* __restrict__ out,
                                                   int n4) {
    const int i = blockIdx.x * 256 + threadIdx.x;
    if (i < n4) {
        const float4 v = ((const float4*)in)[i];
        union { _Float16 h[4]; uint2 u; } t;
        t.h[0] = (_Float16)v.x; t.h[1] = (_Float16)v.y;
        t.h[2] = (_Float16)v.z; t.h[3] = (_Float16)v.w;
        ((uint2*)out)[i] = t.u;
    }
}

// ---------------------------------------------------------------------------
// f32 (1024x1024) -> transposed f16 (outT[n][k] = in[k][n]); 32x32 LDS tiles.
// Weights become n-major once, so GEMM B-tiles are straight async copies.
// ---------------------------------------------------------------------------
__global__ __launch_bounds__(256) void cvt_transpose_f16(const float* __restrict__ in,
                                                         _Float16* __restrict__ outT) {
    __shared__ _Float16 tile[32][33];
    const int k0 = blockIdx.y * 32;
    const int n0 = blockIdx.x * 32;
    const int tx = threadIdx.x & 31;
    const int ty = threadIdx.x >> 5;  // 8 rows per pass
#pragma unroll
    for (int r = 0; r < 32; r += 8)
        tile[ty + r][tx] = (_Float16)in[(size_t)(k0 + ty + r) * 1024 + n0 + tx];
    __syncthreads();
#pragma unroll
    for (int r = 0; r < 32; r += 8)
        outT[(size_t)(n0 + ty + r) * 1024 + k0 + tx] = tile[tx][ty + r];
}

// ---------------------------------------------------------------------------
// WMMA GEMM: C = A(MxK,f16,row) * B, with B given TRANSPOSED (BT = NxK, row).
// Block 256 thr = 8 waves (2x4); tile 128x256, BK=32; wave tile 64x64.
// Double-buffered LDS, async global->LDS staging, one barrier per k-step.
// M%128==0, N%256==0, K%32==0 guaranteed by caller.
// ---------------------------------------------------------------------------
template <int OUT_F32>
__global__ __launch_bounds__(256) void gemm_wmma(
        const _Float16* __restrict__ A, const _Float16* __restrict__ BT,
        _Float16* __restrict__ Ch, float* __restrict__ Cf,
        const float* __restrict__ bias, int M, int N, int K) {
    __shared__ alignas(16) _Float16 sA[2][128 * 32];
    __shared__ alignas(16) _Float16 sBT[2][256 * 32];
    const int tid  = threadIdx.x;
    const int wave = tid >> 5;
    const int lane = tid & 31;
    const int wm   = wave >> 2;  // 0..1
    const int wn   = wave & 3;   // 0..3
    const size_t bm = (size_t)blockIdx.y * 128;
    const size_t bn = (size_t)blockIdx.x * 256;
    (void)M;

    v8f acc[4][4];
#pragma unroll
    for (int i = 0; i < 4; ++i)
#pragma unroll
        for (int j = 0; j < 4; ++j) acc[i][j] = (v8f)0.0f;

    const int arow = tid >> 1;        // 0..127
    const int aseg = (tid & 1) * 16;  // 0 / 16

    auto load_tiles = [&](int buf, int k0) {
        const _Float16* asrc = A + (bm + arow) * K + k0 + aseg;
        async_copy_b128(asrc,     &sA[buf][arow * 32 + aseg]);
        async_copy_b128(asrc + 8, &sA[buf][arow * 32 + aseg + 8]);
        const _Float16* bsrc = BT + (bn + tid) * K + k0;   // one n-row / thread
        async_copy_b128(bsrc,     &sBT[buf][tid * 32]);
        async_copy_b128(bsrc + 8, &sBT[buf][tid * 32 + 8]);
    };

    const int KT = K >> 5;
    load_tiles(0, 0);
    int cur = 0;
    for (int kt = 0; kt < KT; ++kt) {
        wait_async();
        __syncthreads();
        if (kt + 1 < KT) load_tiles(cur ^ 1, (kt + 1) * 32);

        v16h af[4];
#pragma unroll
        for (int mt = 0; mt < 4; ++mt)
            af[mt] = frag_a(&sA[cur][(wm * 64 + mt * 16) * 32], 32);
#pragma unroll
        for (int nt = 0; nt < 4; ++nt) {
            const v16h bf = frag_b(&sBT[cur][(wn * 64 + nt * 16) * 32], 32);
#pragma unroll
            for (int mt = 0; mt < 4; ++mt)
                acc[mt][nt] = wmma32(af[mt], bf, acc[mt][nt]);
        }
        cur ^= 1;
    }

    const int nl = lane & 15;
    const int hf = lane >> 4;
#pragma unroll
    for (int mt = 0; mt < 4; ++mt) {
#pragma unroll
        for (int nt = 0; nt < 4; ++nt) {
            const size_t col  = bn + wn * 64 + nt * 16 + nl;
            const size_t row0 = bm + wm * 64 + mt * 16 + hf * 8;
#pragma unroll
            for (int r = 0; r < 8; ++r) {
                const float v = acc[mt][nt][r];
                if (OUT_F32) Cf[(row0 + r) * N + col] = v + bias[col];
                else         Ch[(row0 + r) * N + col] = (_Float16)v;
            }
        }
    }
}

// ---------------------------------------------------------------------------
// W_ep[b,h] = rq^T * rk  (64x64, K=G=64), stored TRANSPOSED: WepT[bh][e][d]
// ---------------------------------------------------------------------------
__global__ __launch_bounds__(128) void wep_wmma(const _Float16* __restrict__ RQp,
                                                const _Float16* __restrict__ RKp,
                                                _Float16* __restrict__ WepT) {
    const int bh = blockIdx.x;
    const int b = bh >> 4, h = bh & 15;
    __shared__ alignas(16) _Float16 sQT[64 * 64];  // [d][g]
    __shared__ alignas(16) _Float16 sKT[64 * 64];  // [e][g]
    const int tid = threadIdx.x, wave = tid >> 5, lane = tid & 31;
    const int nl = lane & 15, hf = lane >> 4;
    const int g = tid >> 1;
    const int seg = (tid & 1) * 32;
    {
        union { uint4 q[4]; _Float16 hh[32]; } t;
        const uint4* s0 = (const uint4*)(RQp + ((size_t)(b * 64 + g)) * 1024 + h * 64 + seg);
#pragma unroll
        for (int j = 0; j < 4; ++j) t.q[j] = s0[j];
#pragma unroll
        for (int j = 0; j < 32; ++j) sQT[(seg + j) * 64 + g] = t.hh[j];
        const uint4* s1 = (const uint4*)(RKp + ((size_t)(b * 64 + g)) * 1024 + h * 64 + seg);
#pragma unroll
        for (int j = 0; j < 4; ++j) t.q[j] = s1[j];
#pragma unroll
        for (int j = 0; j < 32; ++j) sKT[(seg + j) * 64 + g] = t.hh[j];
    }
    __syncthreads();
    const int m0 = wave * 16;
    v8f acc[4];
#pragma unroll
    for (int nt = 0; nt < 4; ++nt) acc[nt] = (v8f)0.0f;
#pragma unroll
    for (int ks = 0; ks < 2; ++ks) {
        const int k0 = ks * 32;
        const v16h af = frag_a(sQT + m0 * 64 + k0, 64);
#pragma unroll
        for (int nt = 0; nt < 4; ++nt) {
            const v16h bf = frag_b(sKT + (nt * 16) * 64 + k0, 64);
            acc[nt] = wmma32(af, bf, acc[nt]);
        }
    }
#pragma unroll
    for (int nt = 0; nt < 4; ++nt) {
        const size_t base = (size_t)bh * 4096 + (size_t)(nt * 16 + nl) * 64 + m0 + hf * 8;
#pragma unroll
        for (int r = 0; r < 8; ++r) WepT[base + r] = (_Float16)acc[nt][r];
    }
}

// ---------------------------------------------------------------------------
// Flash attention per (b, h, 64-row q-tile), 4 waves x 16 rows.
// scores = (Q*W_ep) * K^T * 0.125 ; online softmax ; O = P*V
// Output channel-major: aoT[b][h*64+dd][l]  (f16)
// ---------------------------------------------------------------------------
__global__ __launch_bounds__(128) void attn_wmma(const _Float16* __restrict__ Qp,
                                                 const _Float16* __restrict__ Kp,
                                                 const _Float16* __restrict__ Vp,
                                                 const _Float16* __restrict__ WepT,
                                                 _Float16* __restrict__ aoT) {
    const int qt = blockIdx.x;
    const int h  = blockIdx.y;
    const int b  = blockIdx.z;
    __shared__ alignas(16) _Float16 sQ[64 * 64];
    __shared__ alignas(16) _Float16 sWT[64 * 64];
    __shared__ alignas(16) _Float16 sQW[64 * 64];
    __shared__ alignas(16) _Float16 sK[64 * 64];
    __shared__ alignas(16) _Float16 sVT[64 * 64];
    __shared__ alignas(16) _Float16 sP[64 * 64];

    const int tid  = threadIdx.x;
    const int wave = tid >> 5;
    const int lane = tid & 31;
    const int nl = lane & 15;
    const int hf = lane >> 4;
    const int m0 = wave * 16;
    const int row = tid >> 1;
    const int seg = (tid & 1) * 32;

    // stage Q tile + W_ep^T asynchronously
    async_copy_row32(Qp + ((size_t)(b * 1024 + qt * 64 + row)) * 1024 + h * 64 + seg,
                     sQ + row * 64 + seg);
    async_copy_row32(WepT + ((size_t)(b * 16 + h)) * 4096 + tid * 32, sWT + tid * 32);
    wait_async();
    __syncthreads();

    {   // qW = Q @ W_ep  -> f16 strip in sQW
        v8f qw[4];
#pragma unroll
        for (int nt = 0; nt < 4; ++nt) qw[nt] = (v8f)0.0f;
#pragma unroll
        for (int ks = 0; ks < 2; ++ks) {
            const int k0 = ks * 32;
            const v16h af = frag_a(sQ + m0 * 64 + k0, 64);
#pragma unroll
            for (int nt = 0; nt < 4; ++nt) {
                const v16h bf = frag_b(sWT + (nt * 16) * 64 + k0, 64);
                qw[nt] = wmma32(af, bf, qw[nt]);
            }
        }
#pragma unroll
        for (int nt = 0; nt < 4; ++nt)
#pragma unroll
            for (int r = 0; r < 8; ++r)
                sQW[(m0 + hf * 8 + r) * 64 + nt * 16 + nl] = (_Float16)qw[nt][r];
    }
    __syncthreads();

    float mrow[8], lrow[8];
    v8f oacc[4];
#pragma unroll
    for (int r = 0; r < 8; ++r) { mrow[r] = -1.0e30f; lrow[r] = 0.0f; }
#pragma unroll
    for (int nt = 0; nt < 4; ++nt) oacc[nt] = (v8f)0.0f;

    for (int st = 0; st < 16; ++st) {
        __syncthreads();  // prior consumers of sK/sVT finished
        // K tile: async straight copy
        async_copy_row32(Kp + ((size_t)(b * 1024 + st * 64 + row)) * 1024 + h * 64 + seg,
                         sK + row * 64 + seg);
        {   // V tile transposed into sVT[d][s] (register transpose)
            union { uint4 q[4]; _Float16 hh[32]; } t;
            const uint4* src = (const uint4*)(Vp + ((size_t)(b * 1024 + st * 64 + row)) * 1024 + h * 64 + seg);
#pragma unroll
            for (int j = 0; j < 4; ++j) t.q[j] = src[j];
#pragma unroll
            for (int j = 0; j < 32; ++j) sVT[(seg + j) * 64 + row] = t.hh[j];
        }
        wait_async();
        __syncthreads();

        // scores = qW @ K^T * scale
        v8f sc[4];
#pragma unroll
        for (int nt = 0; nt < 4; ++nt) sc[nt] = (v8f)0.0f;
#pragma unroll
        for (int ks = 0; ks < 2; ++ks) {
            const int k0 = ks * 32;
            const v16h af = frag_a(sQW + m0 * 64 + k0, 64);
#pragma unroll
            for (int nt = 0; nt < 4; ++nt) {
                const v16h bf = frag_b(sK + (nt * 16) * 64 + k0, 64);
                sc[nt] = wmma32(af, bf, sc[nt]);
            }
        }
#pragma unroll
        for (int nt = 0; nt < 4; ++nt)
#pragma unroll
            for (int r = 0; r < 8; ++r) sc[nt][r] *= 0.125f;

        // online softmax: 16-lane row reductions within each half-wave
        float nm[8];
#pragma unroll
        for (int r = 0; r < 8; ++r)
            nm[r] = fmaxf(fmaxf(sc[0][r], sc[1][r]), fmaxf(sc[2][r], sc[3][r]));
#pragma unroll
        for (int mk = 1; mk <= 8; mk <<= 1)
#pragma unroll
            for (int r = 0; r < 8; ++r)
                nm[r] = fmaxf(nm[r], __shfl_xor(nm[r], mk, 32));

        float rs[8];
#pragma unroll
        for (int r = 0; r < 8; ++r) {
            const float mn   = fmaxf(mrow[r], nm[r]);
            const float corr = __expf(mrow[r] - mn);
            mrow[r] = mn;
            lrow[r] *= corr;
#pragma unroll
            for (int nt = 0; nt < 4; ++nt) oacc[nt][r] *= corr;
            rs[r] = 0.0f;
        }
#pragma unroll
        for (int nt = 0; nt < 4; ++nt)
#pragma unroll
            for (int r = 0; r < 8; ++r) {
                const float p = __expf(sc[nt][r] - mrow[r]);
                sc[nt][r] = p;
                rs[r] += p;
            }
#pragma unroll
        for (int mk = 1; mk <= 8; mk <<= 1)
#pragma unroll
            for (int r = 0; r < 8; ++r) rs[r] += __shfl_xor(rs[r], mk, 32);
#pragma unroll
        for (int r = 0; r < 8; ++r) lrow[r] += rs[r];

        // P -> wave-private LDS strip, then O += P @ V
#pragma unroll
        for (int nt = 0; nt < 4; ++nt)
#pragma unroll
            for (int r = 0; r < 8; ++r)
                sP[(m0 + hf * 8 + r) * 64 + nt * 16 + nl] = (_Float16)sc[nt][r];

#pragma unroll
        for (int ks = 0; ks < 2; ++ks) {
            const int k0 = ks * 32;
            const v16h af = frag_a(sP + m0 * 64 + k0, 64);
#pragma unroll
            for (int nt = 0; nt < 4; ++nt) {
                const v16h bf = frag_b(sVT + (nt * 16) * 64 + k0, 64);
                oacc[nt] = wmma32(af, bf, oacc[nt]);
            }
        }
    }

    // normalize + channel-major store: aoT[b][h*64+dd][l]
#pragma unroll
    for (int nt = 0; nt < 4; ++nt) {
        const int dd = nt * 16 + nl;
        const size_t base = ((size_t)b * 1024 + h * 64 + dd) * 1024 + qt * 64 + m0 + hf * 8;
#pragma unroll
        for (int r = 0; r < 8; ++r)
            aoT[base + r] = (_Float16)(oacc[nt][r] / lrow[r]);
    }
}

// ---------------------------------------------------------------------------
// Host orchestration
// ---------------------------------------------------------------------------
extern "C" void kernel_launch(void* const* d_in, const int* in_sizes, int n_in,
                              void* d_out, int out_size, void* d_ws, size_t ws_size,
                              hipStream_t stream) {
    (void)in_sizes; (void)n_in; (void)out_size; (void)ws_size;
    const float* queries = (const float*)d_in[0];
    const float* keys    = (const float*)d_in[1];
    const float* values  = (const float*)d_in[2];
    const float* routers = (const float*)d_in[3];
    const float* Wq  = (const float*)d_in[4];
    const float* Wk  = (const float*)d_in[5];
    const float* Wv  = (const float*)d_in[6];
    const float* Wlq = (const float*)d_in[7];
    const float* Wlk = (const float*)d_in[8];
    const float* Wo  = (const float*)d_in[9];
    const float* bo  = (const float*)d_in[10];
    float* out = (float*)d_out;

    const size_t NBL = 8ull * 1024 * 1024;  // B*L*D
    const size_t NR  = 8ull * 64 * 1024;    // B*G*D
    const size_t NW  = 1024ull * 1024;      // D*D

    _Float16* w = (_Float16*)d_ws;
    size_t off = 0;
    _Float16* qh = w + off; off += NBL;
    _Float16* kh = w + off; off += NBL;
    _Float16* vh = w + off; off += NBL;
    _Float16* rh = w + off; off += NR;
    _Float16* wqT  = w + off; off += NW;   // transposed f16 weights
    _Float16* wkT  = w + off; off += NW;
    _Float16* wvT  = w + off; off += NW;
    _Float16* wlqT = w + off; off += NW;
    _Float16* wlkT = w + off; off += NW;
    _Float16* woT  = w + off; off += NW;
    _Float16* Qp  = w + off; off += NBL;
    _Float16* Kp  = w + off; off += NBL;
    _Float16* Vp  = w + off; off += NBL;
    _Float16* RQp = w + off; off += NR;
    _Float16* RKp = w + off; off += NR;
    _Float16* WepT = rh;  // reuse: routers-f16 dead after RQp/RKp GEMMs
    _Float16* aoT  = qh;  // reuse: queries-f16 dead after Qp GEMM

    auto cvt = [&](const float* src, _Float16* dst, size_t n) {
        const int n4 = (int)(n / 4);
        cvt_f32_f16<<<dim3((n4 + 255) / 256), dim3(256), 0, stream>>>(src, dst, n4);
    };
    auto cvtT = [&](const float* src, _Float16* dstT) {
        cvt_transpose_f16<<<dim3(32, 32), dim3(256), 0, stream>>>(src, dstT);
    };
    cvt(queries, qh, NBL);
    cvt(keys,    kh, NBL);
    cvt(values,  vh, NBL);
    cvt(routers, rh, NR);
    cvtT(Wq,  wqT);
    cvtT(Wk,  wkT);
    cvtT(Wv,  wvT);
    cvtT(Wlq, wlqT);
    cvtT(Wlk, wlkT);
    cvtT(Wo,  woT);

    // Projections (f16 out); B operands are pre-transposed weights
    gemm_wmma<0><<<dim3(4, 64), dim3(256), 0, stream>>>(qh, wqT, Qp, nullptr, nullptr, 8192, 1024, 1024);
    gemm_wmma<0><<<dim3(4, 64), dim3(256), 0, stream>>>(kh, wkT, Kp, nullptr, nullptr, 8192, 1024, 1024);
    gemm_wmma<0><<<dim3(4, 64), dim3(256), 0, stream>>>(vh, wvT, Vp, nullptr, nullptr, 8192, 1024, 1024);
    gemm_wmma<0><<<dim3(4, 4),  dim3(256), 0, stream>>>(rh, wlqT, RQp, nullptr, nullptr, 512, 1024, 1024);
    gemm_wmma<0><<<dim3(4, 4),  dim3(256), 0, stream>>>(rh, wlkT, RKp, nullptr, nullptr, 512, 1024, 1024);

    // Per-head bilinear matrix (stored transposed)
    wep_wmma<<<dim3(128), dim3(128), 0, stream>>>(RQp, RKp, WepT);

    // Flash attention, channel-major output
    attn_wmma<<<dim3(16, 16, 8), dim3(128), 0, stream>>>(Qp, Kp, Vp, WepT, aoT);

    // Final GEMM (faithful transposed-merge semantics): out = aoT @ Wo + bo
    gemm_wmma<1><<<dim3(4, 64), dim3(256), 0, stream>>>(aoT, woT, nullptr, out, bo, 8192, 1024, 1024);
}
